// CausalLinearTransformerEncoder_7267084665425
// MI455X (gfx1250) — matrix operand order repaired and checked by reading, not compiled, not run on hardware
//
#include <hip/hip_runtime.h>
#include <hip/hip_bf16.h>
#include <stdint.h>

// Problem constants (from reference)
#define NBATCH   2
#define SEQL     2048
#define DMODEL   512
#define NHEAD    8
#define DHEAD    64
#define FFF      2048
#define NLAYERS  2
#define MROWS    (NBATCH * SEQL)

typedef __attribute__((ext_vector_type(16))) __bf16 v16bf;
typedef __attribute__((ext_vector_type(8)))  float  v8f;
typedef __attribute__((__vector_size__(16))) int    v4i;   // builtin's pointee type

union BF16Frag { uint32_t u[8]; v16bf v; };

enum { ACT_NONE = 0, ACT_ELU1 = 1, ACT_GELU = 2 };

#define AS1 __attribute__((address_space(1)))
#define AS3 __attribute__((address_space(3)))

#if __has_builtin(__builtin_amdgcn_global_load_async_to_lds_b128)
#define HAS_ASYNC 1
#else
#define HAS_ASYNC 0
#endif

__device__ __forceinline__ uint16_t f2bf(float f) {
    uint32_t u = __float_as_uint(f);
    return (uint16_t)((u + 0x7FFFu + ((u >> 16) & 1u)) >> 16);  // RNE
}

__device__ __forceinline__ float gelu_tanh(float x) {
    // jax.nn.gelu default (approximate=True)
    float x3 = x * x * x;
    float t = tanhf(0.7978845608028654f * (x + 0.044715f * x3));
    return 0.5f * x * (1.0f + t);
}

__device__ __forceinline__ void wait_async0() {
#if HAS_ASYNC
#if __has_builtin(__builtin_amdgcn_s_wait_asynccnt)
    __builtin_amdgcn_s_wait_asynccnt(0);
#else
    asm volatile("s_wait_asynccnt 0x0" ::: "memory");
#endif
#endif
}

// Stage one 128x32 bf16 A tile and one 128x32 bf16 B tile into LDS.
// Both operands are [rows, K] bf16 row-major, so each (row, 16B-chunk) is a
// contiguous b128 transfer -> per-lane GLOBAL_LOAD_ASYNC_TO_LDS_B128.
// LDS row stride = 24 dwords (16 data + 8 pad, keeps 16B alignment).
__device__ __forceinline__ void stage_tiles(
    const uint16_t* __restrict__ Ab, const uint16_t* __restrict__ Bb, int K,
    uint32_t* la, uint32_t* lb, int tid)
{
#pragma unroll
    for (int i = 0; i < 2; ++i) {
        int c   = tid + i * 256;          // 0..511 chunks per tile
        int row = c >> 2, q4 = c & 3;
        const uint16_t* gA = Ab + (size_t)row * K + q4 * 8;
        const uint16_t* gB = Bb + (size_t)row * K + q4 * 8;
        uint32_t* pA = la + row * 24 + q4 * 4;
        uint32_t* pB = lb + row * 24 + q4 * 4;
#if HAS_ASYNC
        __builtin_amdgcn_global_load_async_to_lds_b128((AS1 v4i*)gA, (AS3 v4i*)pA, 0, 0);
        __builtin_amdgcn_global_load_async_to_lds_b128((AS1 v4i*)gB, (AS3 v4i*)pB, 0, 0);
#else
        *(uint4*)pA = *(const uint4*)gA;
        *(uint4*)pB = *(const uint4*)gB;
#endif
    }
}

// ---------------------------------------------------------------------------
// Tiled GEMM: C[M,N] = act(A[M,K] @ W[K,N] + bias) (+ residual)
// A: bf16 [M,K] row-major. Wt: bf16 [N,K] row-major (pre-transposed weights).
// 128x128 tile / workgroup, 8 waves (wave tile 32x64 = 2x4 WMMA accums),
// K stepped by 32, LDS double-buffered with async global->LDS staging.
// ---------------------------------------------------------------------------
__global__ __launch_bounds__(256) void gemm_bf16_wmma(
    const uint16_t* __restrict__ A, const uint16_t* __restrict__ Wt,
    const float* __restrict__ bias,
    float* __restrict__ Cf,            // fp32 output or nullptr
    uint16_t* __restrict__ Cb,         // bf16 output or nullptr
    int K, int Nout, int act,
    const float* __restrict__ rowscale,    // x_mask row scale (ELU1) or nullptr
    const float* __restrict__ residual)    // fp32 residual or nullptr
{
    __shared__ uint32_t lds[2][6144];  // [buf][ A:128*24 | B:128*24 ] = 48KB

    const int tid   = threadIdx.x;
    const int lane  = tid & 31;
    const int wid   = tid >> 5;
    const int waveM = wid >> 1;        // 0..3 -> 32 rows each
    const int waveN = wid & 1;         // 0..1 -> 64 cols each
    const int ml    = lane & 15;
    const int hi    = lane >> 4;       // lane half selects K sub-range
    const int mBase = blockIdx.y * 128;
    const int nBase = blockIdx.x * 128;

    const uint16_t* Abase = A  + (size_t)mBase * K;
    const uint16_t* Bbase = Wt + (size_t)nBase * K;

    v8f acc[2][4];
    const v8f vzero = {0.f, 0.f, 0.f, 0.f, 0.f, 0.f, 0.f, 0.f};
#pragma unroll
    for (int mi = 0; mi < 2; ++mi)
#pragma unroll
        for (int ni = 0; ni < 4; ++ni) acc[mi][ni] = vzero;

    // prologue: fill buffer 0
    stage_tiles(Abase, Bbase, K, &lds[0][0], &lds[0][3072], tid);
    wait_async0();
    __syncthreads();

    int cur = 0;
    for (int kc = 0; kc < K; kc += 32) {
        // kick off async staging of the next K-chunk into the other buffer
        if (kc + 32 < K)
            stage_tiles(Abase + kc + 32, Bbase + kc + 32, K,
                        &lds[cur ^ 1][0], &lds[cur ^ 1][3072], tid);

        const uint32_t* la = &lds[cur][0];
        const uint32_t* lb = &lds[cur][3072];

        BF16Frag a[2], b[4];
#pragma unroll
        for (int mi = 0; mi < 2; ++mi) {
            int m  = waveM * 32 + mi * 16 + ml;
            int kp = hi ? 4 : 0;                     // lanes16-31: K+8
#pragma unroll
            for (int j = 0; j < 4; ++j) {
                a[mi].u[j]     = la[m * 24 + kp + j];      // K {0..7|8..15}
                a[mi].u[4 + j] = la[m * 24 + 8 + kp + j];  // K +16
            }
        }
#pragma unroll
        for (int ni = 0; ni < 4; ++ni) {
            int n  = waveN * 64 + ni * 16 + ml;
            int kp = hi ? 4 : 0;                     // lanes16-31: K+8
#pragma unroll
            for (int j = 0; j < 8; ++j) b[ni].u[j] = lb[n * 24 + kp + j + (j & 4)];
        }

#pragma unroll
        for (int mi = 0; mi < 2; ++mi)
#pragma unroll
            for (int ni = 0; ni < 4; ++ni)
                acc[mi][ni] = __builtin_amdgcn_wmma_f32_16x16x32_bf16(
                    false, a[mi].v, false, b[ni].v,
                    (short)0, acc[mi][ni], false, false);

        wait_async0();       // own async copies into cur^1 landed
        __syncthreads();     // all waves done copying + done reading cur
        cur ^= 1;
    }

    // ---- epilogue: bias + activation (+row scale) + residual ----
#pragma unroll
    for (int mi = 0; mi < 2; ++mi) {
#pragma unroll
        for (int ni = 0; ni < 4; ++ni) {
#pragma unroll
            for (int r = 0; r < 8; ++r) {
                int row = mBase + waveM * 32 + mi * 16 + r + hi * 8;
                int col = nBase + waveN * 64 + ni * 16 + ml;
                float v = acc[mi][ni][r] + bias[col];
                if (act == ACT_ELU1) {
                    v = v > 0.f ? v + 1.f : __expf(v);    // elu(x)+1
                    if (rowscale) v *= rowscale[row];     // x_mask for K
                } else if (act == ACT_GELU) {
                    v = gelu_tanh(v);
                }
                size_t off = (size_t)row * Nout + col;
                if (residual) v += residual[off];
                if (Cf) Cf[off] = v;
                if (Cb) Cb[off] = f2bf(v);
            }
        }
    }
}

// ---------------------------------------------------------------------------
// Causal linear-attention recurrence. One 64-thread block per (batch, head).
// Thread dv owns column dv of the 64x64 KV state in registers; emits bf16 O.
// ---------------------------------------------------------------------------
__global__ __launch_bounds__(64) void linattn_scan(
    const float* __restrict__ Qf, const float* __restrict__ Kf,
    const float* __restrict__ V,  uint16_t* __restrict__ Ob)
{
    __shared__ float Qs[64], Ks[64], Kc[64];
    const int tid = threadIdx.x;
    const int n = blockIdx.x / NHEAD, h = blockIdx.x % NHEAD;

    float S[64];
#pragma unroll
    for (int j = 0; j < 64; ++j) S[j] = 0.f;
    Kc[tid] = 0.f;

    for (int t = 0; t < SEQL; ++t) {
        __syncthreads();
        size_t idx = ((size_t)(n * SEQL + t)) * DMODEL + h * DHEAD + tid;
        float q = Qf[idx], k = Kf[idx], v = V[idx];
        Qs[tid] = q; Ks[tid] = k; Kc[tid] += k;     // inclusive K cumsum
        __syncthreads();

        float o = 0.f, z = 1e-6f;                   // EPS_ATTN
#pragma unroll
        for (int j = 0; j < 64; ++j) {
            S[j] += Ks[j] * v;                      // inclusive KV prefix
            o    += Qs[j] * S[j];
            z    += Qs[j] * Kc[j];
        }
        Ob[idx] = f2bf(o / z);
    }
}

// ---------------------------------------------------------------------------
// Row LayerNorm over D=512: one wave32 per row; optional bf16 shadow output.
// ---------------------------------------------------------------------------
__global__ __launch_bounds__(256) void layernorm_rows(
    const float* __restrict__ X, const float* __restrict__ g,
    const float* __restrict__ b, float* __restrict__ Y,
    uint16_t* __restrict__ Ybf)
{
    const int lane = threadIdx.x & 31;
    const int row  = blockIdx.x * 8 + (threadIdx.x >> 5);
    const float* xr = X + (size_t)row * DMODEL;

    float vals[16];
    float s = 0.f, ss = 0.f;
#pragma unroll
    for (int i = 0; i < 16; ++i) {
        float f = xr[lane + 32 * i];
        vals[i] = f; s += f; ss += f * f;
    }
#pragma unroll
    for (int off = 16; off > 0; off >>= 1) {
        s  += __shfl_xor(s,  off, 32);
        ss += __shfl_xor(ss, off, 32);
    }
    float mean = s * (1.f / DMODEL);
    float var  = ss * (1.f / DMODEL) - mean * mean;
    float inv  = rsqrtf(var + 1e-5f);               // EPS_LN

    float* yr = Y + (size_t)row * DMODEL;
#pragma unroll
    for (int i = 0; i < 16; ++i) {
        int c = lane + 32 * i;
        float r = (vals[i] - mean) * inv * g[c] + b[c];
        yr[c] = r;
        if (Ybf) Ybf[(size_t)row * DMODEL + c] = f2bf(r);
    }
}

// fp32 -> bf16 elementwise
__global__ __launch_bounds__(256) void cvt_f32_bf16(
    const float* __restrict__ X, uint16_t* __restrict__ Y, int n)
{
    int i = blockIdx.x * 256 + threadIdx.x;
    if (i < n) Y[i] = f2bf(X[i]);
}

// W fp32 [K,N] -> Wt bf16 [N,K] (transpose + convert), coalesced K-writes
__global__ __launch_bounds__(256) void transpose_cvt_bf16(
    const float* __restrict__ W, uint16_t* __restrict__ Wt, int K, int N)
{
    int id = blockIdx.x * 256 + threadIdx.x;
    int k = id % K, n = id / K;
    Wt[(size_t)n * K + k] = f2bf(W[(size_t)k * N + n]);
}

// ---------------------------------------------------------------------------
static inline void gemm(const uint16_t* A, const uint16_t* Wt, const float* bias,
                        float* Cf, uint16_t* Cb, int K, int Nout, int act,
                        const float* rowscale, const float* residual,
                        hipStream_t stream)
{
    dim3 grid(Nout / 128, MROWS / 128);
    gemm_bf16_wmma<<<grid, 256, 0, stream>>>(A, Wt, bias, Cf, Cb, K, Nout, act,
                                             rowscale, residual);
}

extern "C" void kernel_launch(void* const* d_in, const int* in_sizes, int n_in,
                              void* d_out, int out_size, void* d_ws, size_t ws_size,
                              hipStream_t stream)
{
    (void)in_sizes; (void)n_in; (void)out_size; (void)ws_size;

    const float* x    = (const float*)d_in[0];     // [N,L,D]
    const float* mask = (const float*)d_in[1];     // [N,L,1] -> per-row scale

    const size_t MD = (size_t)MROWS * DMODEL;      // 2,097,152

    // workspace: fp32 region then bf16 region
    float* ws   = (float*)d_ws;
    float* xcur = ws;                              // [M,D] fp32 activations
    float* qf   = ws + 1 * MD;
    float* kf   = ws + 2 * MD;
    float* vf   = ws + 3 * MD;
    float* t0   = ws + 4 * MD;                     // pre-LN sums
    uint16_t* hb = (uint16_t*)(ws + 5 * MD);
    uint16_t* xbf = hb;                            // [M,D] bf16 activations
    uint16_t* obf = hb + 1 * MD;                   // [M,D] bf16 attn output
    uint16_t* hbf = hb + 2 * MD;                   // [M,F] bf16 ffn hidden
    uint16_t* wtb = hb + 6 * MD;                   // bf16 transposed weights

    // per-layer transposed-weight slots
    const size_t SZ_DD = (size_t)DMODEL * DMODEL;
    const size_t SZ_DF = (size_t)DMODEL * FFF;
    uint16_t* wT[NLAYERS][6];
    {
        uint16_t* p = wtb;
        for (int l = 0; l < NLAYERS; ++l) {
            for (int w = 0; w < 4; ++w) { wT[l][w] = p; p += SZ_DD; }  // q,k,v,o
            wT[l][4] = p; p += SZ_DF;                                  // w1
            wT[l][5] = p; p += SZ_DF;                                  // w2
        }
    }

    // seed activations (fp32 copy + bf16 shadow)
    (void)hipMemcpyAsync(xcur, x, MD * sizeof(float), hipMemcpyDeviceToDevice, stream);
    cvt_f32_bf16<<<(int)(MD / 256), 256, 0, stream>>>(x, xbf, (int)MD);

    // prep all weights: fp32 [K,N] -> bf16 [N,K]
    for (int l = 0; l < NLAYERS; ++l) {
        void* const* p = &d_in[2 + l * 16];
        const float* wsrc[6] = {(const float*)p[0], (const float*)p[2],
                                (const float*)p[4], (const float*)p[6],
                                (const float*)p[8], (const float*)p[10]};
        const int kk[6] = {DMODEL, DMODEL, DMODEL, DMODEL, DMODEL, FFF};
        const int nn[6] = {DMODEL, DMODEL, DMODEL, DMODEL, FFF, DMODEL};
        for (int w = 0; w < 6; ++w) {
            int total = kk[w] * nn[w];
            transpose_cvt_bf16<<<total / 256, 256, 0, stream>>>(
                wsrc[w], wT[l][w], kk[w], nn[w]);
        }
    }

    for (int layer = 0; layer < NLAYERS; ++layer) {
        void* const* p = &d_in[2 + layer * 16];
        const float* bq  = (const float*)p[1];
        const float* bk  = (const float*)p[3];
        const float* bv  = (const float*)p[5];
        const float* bo  = (const float*)p[7];
        const float* b1  = (const float*)p[9];
        const float* b2  = (const float*)p[11];
        const float* g1  = (const float*)p[12];
        const float* be1 = (const float*)p[13];
        const float* g2  = (const float*)p[14];
        const float* be2 = (const float*)p[15];

        // Q/K/V projections with fused feature map (phi = elu+1; K masked)
        gemm(xbf, wT[layer][0], bq, qf, nullptr, DMODEL, DMODEL, ACT_ELU1, nullptr, nullptr, stream);
        gemm(xbf, wT[layer][1], bk, kf, nullptr, DMODEL, DMODEL, ACT_ELU1, mask,    nullptr, stream);
        gemm(xbf, wT[layer][2], bv, vf, nullptr, DMODEL, DMODEL, ACT_NONE, nullptr, nullptr, stream);

        // causal linear-attention scan -> bf16 O
        linattn_scan<<<NBATCH * NHEAD, 64, 0, stream>>>(qf, kf, vf, obf);

        // output projection + residual, then LN1 (fp32 + bf16 shadow)
        gemm(obf, wT[layer][3], bo, t0, nullptr, DMODEL, DMODEL, ACT_NONE, nullptr, xcur, stream);
        layernorm_rows<<<MROWS / 8, 256, 0, stream>>>(t0, g1, be1, xcur, xbf);

        // FFN: gelu(x@w1+b1) kept bf16-only; @w2 + b2 + residual; LN2
        gemm(xbf, wT[layer][4], b1, nullptr, hbf, DMODEL, FFF, ACT_GELU, nullptr, nullptr, stream);
        gemm(hbf, wT[layer][5], b2, t0, nullptr, FFF, DMODEL, ACT_NONE, nullptr, xcur, stream);

        if (layer == NLAYERS - 1)
            layernorm_rows<<<MROWS / 8, 256, 0, stream>>>(t0, g2, be2, (float*)d_out, nullptr);
        else
            layernorm_rows<<<MROWS / 8, 256, 0, stream>>>(t0, g2, be2, xcur, xbf);
    }
}
